// Attention_49976239456487
// MI455X (gfx1250) — matrix operand order repaired and checked by reading, not compiled, not run on hardware
//
#include <hip/hip_runtime.h>

// ---------------------------------------------------------------------------
// CDNA5 (gfx1250) causal multi-head attention, f16 WMMA pipeline
// with TDM (tensor_load_to_lds) double-buffered weight staging.
// ---------------------------------------------------------------------------

typedef _Float16 f16;
typedef __attribute__((ext_vector_type(4)))  _Float16     v4h;
typedef __attribute__((ext_vector_type(8)))  _Float16     v8h;
typedef __attribute__((ext_vector_type(16))) _Float16     v16h;
typedef __attribute__((ext_vector_type(8)))  float        v8f;
typedef __attribute__((ext_vector_type(4)))  unsigned int v4u;
typedef __attribute__((ext_vector_type(8)))  int          v8i;
typedef __attribute__((ext_vector_type(4)))  int          v4i;

#define WMMA_F32_F16(a, b, c) \
  __builtin_amdgcn_wmma_f32_16x16x32_f16(false, (a), false, (b), (short)0, (c), false, false)

#if defined(__has_builtin)
#if __has_builtin(__builtin_amdgcn_tensor_load_to_lds) && \
    __has_builtin(__builtin_amdgcn_s_wait_tensorcnt)
#define HAVE_TDM 1
#endif
#endif
#ifndef HAVE_TDM
#define HAVE_TDM 0
#endif

constexpr int B = 8, S = 1024, E = 1024, H = 16, D = 64;
constexpr int BS = B * S, HD = H * D;

// --- helpers ---------------------------------------------------------------

__device__ __forceinline__ v16h cat8(v8h lo, v8h hi) {
  return __builtin_shufflevector(lo, hi, 0, 1, 2, 3, 4, 5, 6, 7,
                                         8, 9, 10, 11, 12, 13, 14, 15);
}

// A-fragment (16x32 f16): halves 0..15 = K[hs*16..+15] (contiguous)
__device__ __forceinline__ v16h load_a(const f16* row, int hs) {
  v8h lo = *(const v8h*)(row + hs * 16);
  v8h hi = *(const v8h*)(row + hs * 16 + 8);
  return cat8(lo, hi);
}

// B-fragment (32x16 f16): 16 contiguous K values for column n = lane&15
__device__ __forceinline__ v16h load_b(const f16* p) { return *(const v16h*)p; }

#if HAVE_TDM
// Issue a TDM 2D tile load: rows x rowlen f16 elements, row pitch
// `stride_elems` in global, packed [rows][rowlen] at dstLds.
// D# per CDNA5 ISA (group0 128b, group1 256b).
// 6-arg clang-23 builtin: (v4u g0, v8i g1, v4i g2, v4i g3, v8i g4, i32 cpol).
__device__ __forceinline__ void tdm_issue(f16* dstLds, const f16* src,
                                          unsigned rows, unsigned rowlen,
                                          unsigned stride_elems) {
  const unsigned long long ga = (unsigned long long)(size_t)src;
  const unsigned lds_off = (unsigned)(size_t)dstLds;  // low 32 bits = LDS offset
  v4u g0;
  g0[0] = 1u;                                         // count=1, user descriptor
  g0[1] = lds_off;                                    // lds_addr (bytes)
  g0[2] = (unsigned)(ga & 0xffffffffull);             // global_addr[31:0]
  g0[3] = (unsigned)((ga >> 32) & 0x01ffffffull)      // global_addr[56:32]
        | (2u << 30);                                 // type=2 ("image")
  v8i g1;
  g1[0] = (int)(1u << 16);                            // data_size=1 (2B), wg_mask=0
  g1[1] = (int)((rowlen & 0xffffu) << 16);            // tensor_dim0[15:0]
  g1[2] = (int)(((rowlen >> 16) & 0xffffu)            // tensor_dim0[31:16]
        | ((rows & 0xffffu) << 16));                  // tensor_dim1[15:0]
  g1[3] = (int)(((rows >> 16) & 0xffffu)              // tensor_dim1[31:16]
        | ((rowlen & 0xffffu) << 16));                // tile_dim0
  g1[4] = (int)(rows & 0xffffu);                      // tile_dim1 (tile_dim2=0)
  g1[5] = (int)stride_elems;                          // tensor_dim0_stride[31:0]
  g1[6] = 0;                                          // stride0 hi / stride1 lo
  g1[7] = 0;
  const v4i g2 = {0, 0, 0, 0};                        // 2D tensor: groups 2/3 unused
  const v4i g3 = {0, 0, 0, 0};
  const v8i g4 = {0, 0, 0, 0, 0, 0, 0, 0};            // extended group (unused)
  __builtin_amdgcn_tensor_load_to_lds(g0, g1, g2, g3, g4, 0);
}
#endif

// Stage a [rows][rowlen] f16 tile into LDS. TDM path: wave 0 issues the DMA
// (caller later calls stage_wait + barrier). Fallback: cooperative copy.
__device__ __forceinline__ void stage_tile(f16* dstLds, const f16* src,
                                           int rows, int rowlen, int stride,
                                           int tid, int nthr) {
#if HAVE_TDM
  (void)nthr;
  if (tid < 32) tdm_issue(dstLds, src, (unsigned)rows, (unsigned)rowlen,
                          (unsigned)stride);
#else
  for (int i = tid; i < rows * rowlen; i += nthr)
    dstLds[i] = src[(size_t)(i / rowlen) * stride + (i % rowlen)];
#endif
}

__device__ __forceinline__ void stage_wait(int tid) {
#if HAVE_TDM
  if (tid < 32) __builtin_amdgcn_s_wait_tensorcnt(0);
#else
  (void)tid;
#endif
}

// --- conversion / transpose kernels ----------------------------------------

__global__ void k_cvt4(const float* __restrict__ in, f16* __restrict__ out, int n4) {
  for (int i = blockIdx.x * blockDim.x + threadIdx.x; i < n4;
       i += gridDim.x * blockDim.x) {
    const float4 v = ((const float4*)in)[i];
    v4h o;
    o[0] = (f16)v.x; o[1] = (f16)v.y; o[2] = (f16)v.z; o[3] = (f16)v.w;
    ((v4h*)out)[i] = o;
  }
}

// [batch][R][C] f32 -> [batch][C][R] f16 (column-major for WMMA B operand)
__global__ void k_transpose(const float* __restrict__ in, f16* __restrict__ out,
                            int batch, int R, int C) {
  int i = blockIdx.x * blockDim.x + threadIdx.x;
  int total = batch * R * C;
  if (i >= total) return;
  int c = i % C, r = (i / C) % R, bb = i / (C * R);
  out[((size_t)bb * C + c) * R + r] = (f16)in[i];
}

// --- QKV projection: per (b,h): [S x E] @ [E x D] --------------------------
// B operand (weights) staged through LDS by TDM, double buffered; A operand
// software double-buffered in registers.
// out: vmode==0 -> [B,H,S,D] ; vmode==1 -> [B,H,D,S] (V transposed)

__global__ __launch_bounds__(256) void k_qkv(const f16* __restrict__ xh,  // [BS,E]
                                             const f16* __restrict__ wt,  // [H,D,E]
                                             f16* __restrict__ out, int vmode) {
  __shared__ f16 Wbuf[2][D * 32];  // two 4KB B-tiles: [64 d][32 k]
  const int tid = threadIdx.x;
  const int lane = tid & 31, wave = tid >> 5;
  const int m = lane & 15, hs = lane >> 4;
  const int bh = blockIdx.y, b = bh / H, h = bh % H;
  const int row0 = blockIdx.x * 128 + wave * 16;
  constexpr int NK = E / 32;

  const f16* arow  = xh + (size_t)(b * S + row0 + m) * E;
  const f16* wbase = wt + (size_t)h * D * E;

  stage_tile(&Wbuf[0][0], wbase, D, 32, E, tid, 256);
  stage_wait(tid);
  __syncthreads();

  v8f acc[4] = {};
  v16h a_cur = load_a(arow, hs);
  for (int ks = 0; ks < NK; ++ks) {
    const bool more = (ks + 1 < NK);
    if (more)  // kick next weight tile while computing on the current one
      stage_tile(&Wbuf[(ks + 1) & 1][0], wbase + (ks + 1) * 32, D, 32, E, tid, 256);
    v16h a_nxt = a_cur;
    if (more) a_nxt = load_a(arow + (ks + 1) * 32, hs);  // overlap A load
    const f16* cur = &Wbuf[ks & 1][0];
#pragma unroll
    for (int t = 0; t < 4; ++t) {
      const f16* bp = cur + (t * 16 + m) * 32 + hs * 16;
      acc[t] = WMMA_F32_F16(a_cur, load_b(bp), acc[t]);
    }
    a_cur = a_nxt;
    if (more) stage_wait(tid);
    __syncthreads();
  }

  if (!vmode) {  // q/k: [B,H,S,D]
    f16* obase = out + (size_t)bh * S * D;
#pragma unroll
    for (int r = 0; r < 8; ++r) {
      f16* orow = obase + (size_t)(row0 + r + 8 * hs) * D;
#pragma unroll
      for (int t = 0; t < 4; ++t) orow[t * 16 + m] = (f16)acc[t][r];
    }
  } else {       // v: [B,H,D,S] -- lane's 8 rows are contiguous in S
    f16* obase = out + (size_t)bh * D * S;
#pragma unroll
    for (int t = 0; t < 4; ++t) {
      v8h pk;
#pragma unroll
      for (int r = 0; r < 8; ++r) pk[r] = (f16)acc[t][r];
      *(v8h*)(obase + (size_t)(t * 16 + m) * S + row0 + 8 * hs) = pk;
    }
  }
}

// --- Flash attention: one wave = 16 queries, key tiles of 32 ---------------

__global__ __launch_bounds__(128) void k_flash(const f16* __restrict__ q,   // [B,H,S,D]
                                               const f16* __restrict__ k,   // [B,H,S,D]
                                               const f16* __restrict__ vt,  // [B,H,D,S]
                                               f16* __restrict__ o) {       // [BS,HD]
  __shared__ f16 Plds[4][16][32];  // per-wave P-tile staging (C-layout -> A-layout)
  const int lane = threadIdx.x & 31, wave = threadIdx.x >> 5;
  const int m = lane & 15, hs = lane >> 4;
  const int bh = blockIdx.y, b = bh / H, h = bh % H;
  const int q0 = blockIdx.x * 64 + wave * 16;

  const f16* qp = q  + (size_t)bh * S * D;
  const f16* kp = k  + (size_t)bh * S * D;
  const f16* vp = vt + (size_t)bh * D * S;

  const f16* qrow = qp + (size_t)(q0 + m) * D;
  const v16h aq0 = load_a(qrow, hs);
  const v16h aq1 = load_a(qrow + 32, hs);

  float mrow[8], lrow[8];
  v8f oacc[4] = {};
#pragma unroll
  for (int r = 0; r < 8; ++r) { mrow[r] = -1e30f; lrow[r] = 0.f; }

  const float scale = 0.125f;  // 1/sqrt(D)

  for (int kb = 0; kb <= q0 + 15; kb += 32) {  // causal: skip tiles past diagonal
    // prefetch next key tile rows into L2/WGP$ (global_prefetch_b8)
    __builtin_prefetch(kp + (size_t)(kb + 32 + m) * D, 0, 0);

    // scores S = Q K^T for 32 keys (two 16-col WMMA tiles)
    v8f c0 = {}, c1 = {};
    {
      const f16* kr0 = kp + (size_t)(kb + m) * D + hs * 16;
      c0 = WMMA_F32_F16(aq0, load_b(kr0), c0);
      c0 = WMMA_F32_F16(aq1, load_b(kr0 + 32), c0);
      const f16* kr1 = kp + (size_t)(kb + 16 + m) * D + hs * 16;
      c1 = WMMA_F32_F16(aq0, load_b(kr1), c1);
      c1 = WMMA_F32_F16(aq1, load_b(kr1 + 32), c1);
    }

    const int col0 = kb + m, col1 = kb + 16 + m;
    float p0[8], p1[8], mx[8];
#pragma unroll
    for (int r = 0; r < 8; ++r) {
      const int row = q0 + r + 8 * hs;       // C-layout row for this lane half
      float s0 = c0[r] * scale; if (col0 > row) s0 = -1e30f;
      float s1 = c1[r] * scale; if (col1 > row) s1 = -1e30f;
      p0[r] = s0; p1[r] = s1;
      float t = fmaxf(s0, s1);               // row-max across the 16-lane half
      t = fmaxf(t, __shfl_xor(t, 1));
      t = fmaxf(t, __shfl_xor(t, 2));
      t = fmaxf(t, __shfl_xor(t, 4));
      t = fmaxf(t, __shfl_xor(t, 8));
      mx[r] = t;
    }
#pragma unroll
    for (int r = 0; r < 8; ++r) {
      const float mn    = fmaxf(mrow[r], mx[r]);
      const float alpha = __expf(mrow[r] - mn);
      mrow[r] = mn;
      const float e0 = __expf(p0[r] - mn);
      const float e1 = __expf(p1[r] - mn);
      float rs = e0 + e1;                    // row-sum across the half
      rs += __shfl_xor(rs, 1);
      rs += __shfl_xor(rs, 2);
      rs += __shfl_xor(rs, 4);
      rs += __shfl_xor(rs, 8);
      lrow[r] = lrow[r] * alpha + rs;
#pragma unroll
      for (int t2 = 0; t2 < 4; ++t2) oacc[t2][r] *= alpha;
      Plds[wave][r + 8 * hs][m]      = (f16)e0;
      Plds[wave][r + 8 * hs][16 + m] = (f16)e1;
    }
    asm volatile("s_wait_dscnt 0" ::: "memory");  // LDS RAW fence (same wave)
    const v16h ap = load_a(&Plds[wave][m][0], hs);
#pragma unroll
    for (int t = 0; t < 4; ++t) {
      const f16* vrow = vp + (size_t)(t * 16 + m) * S + kb + hs * 16;
      oacc[t] = WMMA_F32_F16(ap, load_b(vrow), oacc[t]);
    }
  }

  // normalize and store directly into concat-heads layout [BS, H*D]
#pragma unroll
  for (int r = 0; r < 8; ++r) {
    const float inv = 1.0f / lrow[r];
    const int row = q0 + r + 8 * hs;
    f16* orow = o + (size_t)(b * S + row) * HD + h * D;
#pragma unroll
    for (int t = 0; t < 4; ++t) orow[t * 16 + m] = (f16)(oacc[t][r] * inv);
  }
}

// --- output projection: [BS,HD] @ [HD,E] + bo, fp32 out --------------------

__global__ __launch_bounds__(256) void k_proj(const f16* __restrict__ ao,   // [BS,HD]
                                              const f16* __restrict__ wot,  // [E,HD]
                                              const float* __restrict__ bo,
                                              float* __restrict__ out) {    // [BS,E]
  __shared__ f16 Wbuf[2][64 * 32];  // two 4KB B-tiles: [64 e][32 k]
  const int tid = threadIdx.x;
  const int lane = tid & 31, wave = tid >> 5;
  const int m = lane & 15, hs = lane >> 4;
  const int row0 = blockIdx.x * 128 + wave * 16;
  const int col0 = blockIdx.y * 64;
  constexpr int NK = HD / 32;

  const f16* arow  = ao + (size_t)(row0 + m) * HD;
  const f16* wbase = wot + (size_t)col0 * HD;

  stage_tile(&Wbuf[0][0], wbase, 64, 32, HD, tid, 256);
  stage_wait(tid);
  __syncthreads();

  v8f acc[4] = {};
  v16h a_cur = load_a(arow, hs);
  for (int ks = 0; ks < NK; ++ks) {
    const bool more = (ks + 1 < NK);
    if (more)
      stage_tile(&Wbuf[(ks + 1) & 1][0], wbase + (ks + 1) * 32, 64, 32, HD, tid, 256);
    v16h a_nxt = a_cur;
    if (more) a_nxt = load_a(arow + (ks + 1) * 32, hs);
    const f16* cur = &Wbuf[ks & 1][0];
#pragma unroll
    for (int t = 0; t < 4; ++t) {
      const f16* bp = cur + (t * 16 + m) * 32 + hs * 16;
      acc[t] = WMMA_F32_F16(a_cur, load_b(bp), acc[t]);
    }
    a_cur = a_nxt;
    if (more) stage_wait(tid);
    __syncthreads();
  }

#pragma unroll
  for (int t = 0; t < 4; ++t) {
    const int e = col0 + t * 16 + m;
    const float bias = bo[e];
#pragma unroll
    for (int r = 0; r < 8; ++r)
      out[(size_t)(row0 + r + 8 * hs) * E + e] = acc[t][r] + bias;
  }
}

// --- host ------------------------------------------------------------------

extern "C" void kernel_launch(void* const* d_in, const int* in_sizes, int n_in,
                              void* d_out, int out_size, void* d_ws, size_t ws_size,
                              hipStream_t stream) {
  const float* x  = (const float*)d_in[0];
  const float* Wq = (const float*)d_in[1];
  const float* Wk = (const float*)d_in[2];
  const float* Wv = (const float*)d_in[3];
  const float* Wo = (const float*)d_in[4];
  const float* bo = (const float*)d_in[5];
  float* out = (float*)d_out;
  (void)in_sizes; (void)n_in; (void)out_size; (void)ws_size;

  char* ws = (char*)d_ws;
  size_t off = 0;
  auto carve = [&](size_t bytes) -> char* {
    char* p = ws + off;
    off += (bytes + 255) & ~(size_t)255;
    return p;
  };
  f16* xh  = (f16*)carve((size_t)BS * E * 2);        // x in f16
  f16* WqT = (f16*)carve((size_t)H * D * E * 2);     // [H,D,E]
  f16* WkT = (f16*)carve((size_t)H * D * E * 2);
  f16* WvT = (f16*)carve((size_t)H * D * E * 2);
  f16* WoT = (f16*)carve((size_t)E * HD * 2);        // [E,HD]
  f16* qb  = (f16*)carve((size_t)B * H * S * D * 2); // [B,H,S,D]
  f16* kb  = (f16*)carve((size_t)B * H * S * D * 2); // [B,H,S,D]
  f16* vT  = (f16*)carve((size_t)B * H * D * S * 2); // [B,H,D,S]
  f16* ao  = (f16*)carve((size_t)BS * HD * 2);       // [BS,HD]

  k_cvt4<<<2048, 256, 0, stream>>>(x, xh, BS * E / 4);
  k_transpose<<<(H * E * D + 255) / 256, 256, 0, stream>>>(Wq, WqT, H, E, D);
  k_transpose<<<(H * E * D + 255) / 256, 256, 0, stream>>>(Wk, WkT, H, E, D);
  k_transpose<<<(H * E * D + 255) / 256, 256, 0, stream>>>(Wv, WvT, H, E, D);
  k_transpose<<<(HD * E + 255) / 256, 256, 0, stream>>>(Wo, WoT, 1, HD, E);

  dim3 gq(S / 128, B * H);
  k_qkv<<<gq, 256, 0, stream>>>(xh, WqT, qb, 0);
  k_qkv<<<gq, 256, 0, stream>>>(xh, WkT, kb, 0);
  k_qkv<<<gq, 256, 0, stream>>>(xh, WvT, vT, 1);

  k_flash<<<dim3(S / 64, B * H), 128, 0, stream>>>(qb, kb, vT, ao);

  k_proj<<<dim3(BS / 128, E / 64), 256, 0, stream>>>(ao, WoT, bo, out);
}
